// GCN_52896817218206
// MI455X (gfx1250) — matrix operand order repaired and checked by reading, not compiled, not run on hardware
//
#include <hip/hip_runtime.h>

// GCN: 2x GCNConv(+ReLU) + Linear head for MI455X (gfx1250, wave32, WMMA).
//   deg = segsum(ew, col) (+1 self loop); dis = rsqrt(deg)
//   per layer: H = X @ W (WMMA f16->f32), agg[col] += H[row]*dis[row]*ew*dis[col]
//              X' = relu(agg + H*dis^2 + b)   (self-loop folded into epilogue, emitted as f16)
//   out = X'' @ Wfc + bfc
//
// GEMMs are fully compile-time shaped: A is f16 row-major with K padded to 32,
// B is pre-packed into WMMA fragment order so the inner loop is
// 2x b128 (A) + 2x b128 (B) + v_wmma per K-step, no guards, one wait per step.

#define NNODES 100000
#define NEDGES 1600000
#define NCLASS 26
#define HID    128

typedef __attribute__((ext_vector_type(16))) _Float16 v16h;
typedef __attribute__((ext_vector_type(8)))  _Float16 v8h;
typedef __attribute__((ext_vector_type(8)))  float    v8f;

// ---------------- small elementwise kernels ----------------

__global__ void k_fill1(float* __restrict__ p, int n) {
    int i = blockIdx.x * blockDim.x + threadIdx.x;
    if (i < n) p[i] = 1.0f;                      // self-loop weight contribution
}

__global__ void k_deg_accum(const int* __restrict__ col, const float* __restrict__ ew,
                            float* __restrict__ deg, int e) {
    int i = blockIdx.x * blockDim.x + threadIdx.x;
    if (i < e) unsafeAtomicAdd(&deg[col[i]], ew[i]);
}

__global__ void k_rsqrt(float* __restrict__ deg, int n) {
    int i = blockIdx.x * blockDim.x + threadIdx.x;
    if (i < n) {
        float d = deg[i];
        deg[i] = (d > 0.f) ? rsqrtf(d) : 0.f;
    }
}

// x [NNODES, NCLASS] f32 -> xh [NNODES, 32] f16, zero padded cols 26..31
__global__ void k_cvt_pad_x(const float* __restrict__ x, _Float16* __restrict__ xh) {
    int idx = blockIdx.x * blockDim.x + threadIdx.x;
    if (idx >= NNODES * 32) return;
    int node = idx >> 5;
    int c    = idx & 31;
    xh[idx] = (c < NCLASS) ? (_Float16)x[node * NCLASS + c] : (_Float16)0.f;
}

// Pack B[K,N] (f32, row-major) into WMMA fragment order, zero-padded:
//   P[(((kstep*tilesN)+tn)*32 + lane)*16 + e] = B[kstep*32 + (e>>3)*16 + (lane>>4)*8 + (e&7), tn*16 + (lane&15)]
__global__ void k_pack_b(const float* __restrict__ B, _Float16* __restrict__ P,
                         int Kact, int Kpad, int Nact, int tilesN) {
    int idx = blockIdx.x * blockDim.x + threadIdx.x;
    int total = (Kpad / 32) * tilesN * 512;
    if (idx >= total) return;
    int e     = idx & 15;
    int lane  = (idx >> 4) & 31;
    int t     = idx >> 9;
    int tn    = t % tilesN;
    int kstep = t / tilesN;
    int n  = tn * 16 + (lane & 15);
    int kk = kstep * 32 + (e >> 3) * 16 + (lane >> 4) * 8 + (e & 7);
    float v = (kk < Kact && n < Nact) ? B[kk * Nact + n] : 0.f;
    P[idx] = (_Float16)v;
}

// Xh = relu(agg + H*dis^2 + b) emitted as f16 [NNODES, HID]
__global__ void k_combine_relu_f16(const float* __restrict__ agg, const float* __restrict__ H,
                                   const float* __restrict__ dis, const float* __restrict__ bias,
                                   _Float16* __restrict__ Xh) {
    size_t idx = (size_t)blockIdx.x * blockDim.x + threadIdx.x;
    if (idx >= (size_t)NNODES * HID) return;
    int node = (int)(idx >> 7);
    int f    = (int)(idx & (HID - 1));
    float d = dis[node];
    float v = agg[idx] + H[idx] * d * d + bias[f];
    Xh[idx] = (_Float16)(v > 0.f ? v : 0.f);
}

// ---------------- edge scatter: one wave per edge, 4 feats/lane ----------------

__global__ void k_scatter(const float* __restrict__ H, const int* __restrict__ row,
                          const int* __restrict__ col, const float* __restrict__ ew,
                          const float* __restrict__ dis, float* __restrict__ agg) {
    int e    = (blockIdx.x * blockDim.x + threadIdx.x) >> 5;
    int lane = threadIdx.x & 31;
    if (e >= NEDGES) return;
    int r = row[e];
    int c = col[e];
    float norm = dis[r] * ew[e] * dis[c];
    const float4* h4 = (const float4*)(H + (size_t)r * HID);
    float4 v = h4[lane];
    float* dst = agg + (size_t)c * HID + lane * 4;
    unsafeAtomicAdd(dst + 0, v.x * norm);
    unsafeAtomicAdd(dst + 1, v.y * norm);
    unsafeAtomicAdd(dst + 2, v.z * norm);
    unsafeAtomicAdd(dst + 3, v.w * norm);
}

// ---------------- WMMA GEMM ----------------
// C[M,NOUT] = A[M,K](f16) * B(packed f16) (+bias). One wave per 16x16 tile.
// A fragment (ISA 16-bit A 16x32): lane row m = m0+(lane&15); per K-step the lane's
// 16 halfs are two contiguous runs: [k0 + half*8, +8) and [k0 + 16 + half*8, +8).
// B fragment: one contiguous 32-byte packed load.
template <int K, int NOUT, int TILESN, bool BIAS>
__global__ void k_gemm_wmma(const _Float16* __restrict__ A, const _Float16* __restrict__ Bp,
                            float* __restrict__ C, const float* __restrict__ bias) {
    int wave = (int)((blockIdx.x * blockDim.x + threadIdx.x) >> 5);
    int lane = threadIdx.x & 31;
    int tm = wave / TILESN;
    int tn = wave - tm * TILESN;
    if (tm * 16 >= NNODES) return;            // whole wave exits together
    int half = lane >> 4;

    const _Float16* ap = A + (size_t)(tm * 16 + (lane & 15)) * K + half * 8;
    const _Float16* bp = Bp + ((size_t)tn * 32 + lane) * 16;

    v8f acc = {};
#pragma unroll
    for (int ks = 0; ks < K / 32; ++ks) {
        v8h lo = *(const v8h*)(ap + ks * 32);
        v8h hi = *(const v8h*)(ap + ks * 32 + 16);
        v16h a = __builtin_shufflevector(lo, hi, 0, 1, 2, 3, 4, 5, 6, 7,
                                                 8, 9, 10, 11, 12, 13, 14, 15);
        v16h b = *(const v16h*)(bp + (size_t)ks * TILESN * 512);
        acc = __builtin_amdgcn_wmma_f32_16x16x32_f16(
            /*neg_a=*/false, a, /*neg_b=*/false, b,
            /*c_mod=*/(short)0, acc, /*reuse_a=*/false, /*reuse_b=*/false);
    }

    int n = tn * 16 + (lane & 15);
    if (n < NOUT) {
        float* cp = C + (size_t)(tm * 16 + 8 * half) * NOUT + n;
        if constexpr (BIAS) {
            float bv = bias[n];
#pragma unroll
            for (int r = 0; r < 8; ++r) cp[(size_t)r * NOUT] = acc[r] + bv;
        } else {
#pragma unroll
            for (int r = 0; r < 8; ++r) cp[(size_t)r * NOUT] = acc[r];
        }
    }
}

// ---------------- launch ----------------

extern "C" void kernel_launch(void* const* d_in, const int* in_sizes, int n_in,
                              void* d_out, int out_size, void* d_ws, size_t ws_size,
                              hipStream_t stream) {
    (void)in_sizes; (void)n_in; (void)out_size; (void)ws_size;

    const float* x   = (const float*)d_in[0];   // [NNODES, NCLASS]
    const int*   eix = (const int*)d_in[1];     // [2, NEDGES] (int32 per harness)
    const float* ew  = (const float*)d_in[2];   // [NEDGES]
    const float* W1  = (const float*)d_in[3];   // [NCLASS, HID]
    const float* b1  = (const float*)d_in[4];   // [HID]
    const float* W2  = (const float*)d_in[5];   // [HID, HID]
    const float* b2  = (const float*)d_in[6];   // [HID]
    const float* Wfc = (const float*)d_in[7];   // [HID, NCLASS]
    const float* bfc = (const float*)d_in[8];   // [NCLASS]
    float*       out = (float*)d_out;           // [NNODES, NCLASS]

    const int* row = eix;            // sources
    const int* col = eix + NEDGES;   // targets

    // workspace carve-up (~135 MB)
    char* ws = (char*)d_ws;
    size_t off = 0;
    auto carve = [&](size_t bytes) {
        void* p = ws + off;
        off += (bytes + 255) & ~(size_t)255;
        return p;
    };
    float*    dis  = (float*)carve((size_t)NNODES * 4);            // deg -> rsqrt(deg)
    float*    bufA = (float*)carve((size_t)NNODES * HID * 4);      // H (gemm out)
    float*    bufB = (float*)carve((size_t)NNODES * HID * 4);      // agg
    _Float16* bufH = (_Float16*)carve((size_t)NNODES * HID * 2);   // f16 activations
    _Float16* xh   = (_Float16*)carve((size_t)NNODES * 32 * 2);    // padded f16 x
    _Float16* w1p  = (_Float16*)carve((size_t)1 * 8 * 512 * 2);    // packed W1 (K=32)
    _Float16* w2p  = (_Float16*)carve((size_t)4 * 8 * 512 * 2);    // packed W2 (K=128)
    _Float16* wfcp = (_Float16*)carve((size_t)4 * 2 * 512 * 2);    // packed Wfc (K=128,N->32)

    const int BT = 256;

    // input padding + weight packing
    k_cvt_pad_x<<<(NNODES * 32 + BT - 1) / BT, BT, 0, stream>>>(x, xh);
    k_pack_b<<<(1 * 8 * 512 + BT - 1) / BT, BT, 0, stream>>>(W1, w1p, NCLASS, 32, HID, 8);
    k_pack_b<<<(4 * 8 * 512 + BT - 1) / BT, BT, 0, stream>>>(W2, w2p, HID, HID, HID, 8);
    k_pack_b<<<(4 * 2 * 512 + BT - 1) / BT, BT, 0, stream>>>(Wfc, wfcp, HID, HID, NCLASS, 2);

    // degree -> dis = rsqrt(deg)   (deg includes self-loop weight 1.0)
    k_fill1<<<(NNODES + BT - 1) / BT, BT, 0, stream>>>(dis, NNODES);
    k_deg_accum<<<(NEDGES + BT - 1) / BT, BT, 0, stream>>>(col, ew, dis, NEDGES);
    k_rsqrt<<<(NNODES + BT - 1) / BT, BT, 0, stream>>>(dis, NNODES);

    const size_t elemNH  = (size_t)NNODES * HID;
    const int combBlocks = (int)((elemNH + BT - 1) / BT);
    const int scatBlocks = (int)(((size_t)NEDGES * 32 + BT - 1) / BT);
    const int gemmBlocksH  = ((NNODES / 16) * 8 * 32 + BT - 1) / BT;   // N=128
    const int gemmBlocksFC = ((NNODES / 16) * 2 * 32 + BT - 1) / BT;   // N=26->2 tiles

    // ---- layer 1: H1 = xh @ W1 ----
    k_gemm_wmma<32, HID, 8, false><<<gemmBlocksH, BT, 0, stream>>>(xh, w1p, bufA, nullptr);
    hipMemsetAsync(bufB, 0, elemNH * sizeof(float), stream);
    k_scatter<<<scatBlocks, BT, 0, stream>>>(bufA, row, col, ew, dis, bufB);
    k_combine_relu_f16<<<combBlocks, BT, 0, stream>>>(bufB, bufA, dis, b1, bufH);

    // ---- layer 2: H2 = X2 @ W2 ----
    k_gemm_wmma<HID, HID, 8, false><<<gemmBlocksH, BT, 0, stream>>>(bufH, w2p, bufA, nullptr);
    hipMemsetAsync(bufB, 0, elemNH * sizeof(float), stream);
    k_scatter<<<scatBlocks, BT, 0, stream>>>(bufA, row, col, ew, dis, bufB);
    k_combine_relu_f16<<<combBlocks, BT, 0, stream>>>(bufB, bufA, dis, b2, bufH);

    // ---- FC head: out = X3 @ Wfc + bfc ----
    k_gemm_wmma<HID, NCLASS, 2, true><<<gemmBlocksFC, BT, 0, stream>>>(bufH, wfcp, out, bfc);
}
